// BertSelfAttention_10110353015430
// MI455X (gfx1250) — matrix operand order repaired and checked by reading, not compiled, not run on hardware
//
#include <hip/hip_runtime.h>

#define B_SZ     2
#define L_SEQ    2048
#define DM       1024
#define NH       16
#define HD       64

typedef __attribute__((ext_vector_type(16))) __bf16 v16bf;
typedef __attribute__((ext_vector_type(8)))  __bf16 v8bf;
typedef __attribute__((ext_vector_type(8)))  float  v8f;

typedef __attribute__((ext_vector_type(4))) unsigned int u32x4;
typedef __attribute__((ext_vector_type(8))) int          i32x8;
typedef __attribute__((ext_vector_type(4))) int          i32x4;

union AFrag { v16bf v; v8bf h[2]; };

#if defined(__has_builtin)
#if __has_builtin(__builtin_amdgcn_tensor_load_to_lds) && __has_builtin(__builtin_amdgcn_s_wait_tensorcnt)
#define HAVE_TDM 1
#endif
#endif

#ifdef HAVE_TDM
// ---------------------------------------------------------------------------
// Tensor Data Mover: load a 2D tile (tile_x elems of 2B  x  tile_y rows)
// from global (row stride = stride_elems) into LDS at byte offset lds_off.
// D# layout per CDNA5 ISA (groups 0/1; 2D -> rest zero).
// Toolchain uses the 6-arg builtin form (clang-23 / therock-10.0 headers).
// ---------------------------------------------------------------------------
__device__ __forceinline__ void tdm_load_tile_2d(unsigned lds_off, const void* gptr,
                                                 unsigned tile_x, unsigned tile_y,
                                                 unsigned long long stride_elems) {
    const unsigned long long ga = (unsigned long long)(uintptr_t)gptr;
    u32x4 g0;
    g0[0] = 1u;                                        // count=1 (valid), user mode
    g0[1] = lds_off;                                   // lds_addr (bytes)
    g0[2] = (unsigned)ga;                              // global_addr[31:0]
    g0[3] = (unsigned)((ga >> 32) & 0x01ffffffull)     // global_addr[56:32]
          | 0x80000000u;                               // type = 2 ("image")

    const unsigned td0 = (unsigned)stride_elems;       // tensor_dim0 (>= tile_x)
    const unsigned td1 = tile_y;                       // tensor_dim1 (>= tile_y)
    const unsigned s_lo = (unsigned)stride_elems;
    const unsigned s_hi = (unsigned)(stride_elems >> 32) & 0xffffu;
    i32x8 g1;
    g1[0] = 0x00010000;                                // wg_mask=0, data_size=1 (2 bytes)
    g1[1] = (int)((td0 & 0xffffu) << 16);              // barrier_addr=0 | tensor_dim0[15:0]
    g1[2] = (int)((td0 >> 16) | ((td1 & 0xffffu) << 16));
    g1[3] = (int)((td1 >> 16) | (tile_x << 16));       // tensor_dim1[31:16] | tile_dim0
    g1[4] = (int)(tile_y & 0xffffu);                   // tile_dim1 | tile_dim2=0
    g1[5] = (int)s_lo;                                 // tensor_dim0_stride[31:0]
    g1[6] = (int)s_hi;                                 // stride[47:32] | dim1_stride[15:0]=0
    g1[7] = 0;
    const i32x4 gz4 = {0, 0, 0, 0};
    const i32x8 gz8 = {0, 0, 0, 0, 0, 0, 0, 0};
    __builtin_amdgcn_tensor_load_to_lds(g0, g1, gz4, gz4, gz8, 0);
}
#endif

// ---------------------------------------------------------------------------
// fp32 -> bf16 convert
// ---------------------------------------------------------------------------
__global__ void f32_to_bf16(const float* __restrict__ in, __bf16* __restrict__ out, int n) {
    int i = blockIdx.x * 256 + threadIdx.x;
    if (i < n) out[i] = (__bf16)in[i];
}

// ---------------------------------------------------------------------------
// C[M,N] = A[M,K] @ W[N,K]^T   (A, W bf16 K-major; C bf16 or f32)
// block = 256 threads = 8 waves (2 M x 4 N); block tile 128x128, K step 32.
// Data path: TDM tensor_load_to_lds (double buffered) -> ds_load frags -> WMMA.
// ---------------------------------------------------------------------------
template <bool OUT_BF16>
__global__ void __launch_bounds__(256)
wmma_gemm(const __bf16* __restrict__ A, const __bf16* __restrict__ W,
          void* __restrict__ Cout, int M, int N, int K) {
    __shared__ __align__(128) __bf16 ldsA[2][128][32];
    __shared__ __align__(128) __bf16 ldsB[2][128][32];

    const int lane = threadIdx.x & 31;
    const int wave = threadIdx.x >> 5;
    const int wm   = wave >> 2;            // 0..1
    const int wn   = wave & 3;             // 0..3
    const int half = lane >> 4;
    const int l15  = lane & 15;
    const int kbA  = half * 8;
    const int kbB  = half * 16;
    const int bm0  = blockIdx.y * 128;
    const int bn0  = blockIdx.x * 128;

    v8f acc[4][2] = {};

#ifdef HAVE_TDM
    const unsigned ldsAoff = (unsigned)(uintptr_t)(void*)&ldsA[0][0][0];
    const unsigned ldsBoff = (unsigned)(uintptr_t)(void*)&ldsB[0][0][0];
    const unsigned bufBytes = 128 * 32 * 2;

    if (wave == 0) {   // prime buffer 0
        tdm_load_tile_2d(ldsAoff, A + (size_t)bm0 * K, 32, 128, (unsigned long long)K);
        tdm_load_tile_2d(ldsBoff, W + (size_t)bn0 * K, 32, 128, (unsigned long long)K);
    }
#endif

    for (int k0 = 0; k0 < K; k0 += 32) {
        const int cur = (k0 >> 5) & 1;

#ifdef HAVE_TDM
        if (wave == 0) {
            if (k0 + 32 < K) {   // prefetch next pair into the other buffer
                const unsigned off = (unsigned)(cur ^ 1) * bufBytes;
                tdm_load_tile_2d(ldsAoff + off, A + (size_t)bm0 * K + k0 + 32,
                                 32, 128, (unsigned long long)K);
                tdm_load_tile_2d(ldsBoff + off, W + (size_t)bn0 * K + k0 + 32,
                                 32, 128, (unsigned long long)K);
                __builtin_amdgcn_s_wait_tensorcnt(2);  // current pair complete
            } else {
                __builtin_amdgcn_s_wait_tensorcnt(0);
            }
        }
        __syncthreads();
        AFrag a[4], b[2];
#pragma unroll
        for (int i = 0; i < 4; ++i) {
            const __bf16* pa = &ldsA[cur][wm * 64 + i * 16 + l15][0];
            a[i].h[0] = *(const v8bf*)(pa + kbB);
            a[i].h[1] = *(const v8bf*)(pa + kbB + 8);
        }
#pragma unroll
        for (int j = 0; j < 2; ++j) {
            const __bf16* pb = &ldsB[cur][wn * 32 + j * 16 + l15][0];
            b[j].h[0] = *(const v8bf*)(pb + kbB);
            b[j].h[1] = *(const v8bf*)(pb + kbB + 8);
        }
#else
        (void)cur;
        AFrag a[4], b[2];
#pragma unroll
        for (int i = 0; i < 4; ++i) {
            const __bf16* pa = A + (size_t)(bm0 + wm * 64 + i * 16 + l15) * K + k0;
            a[i].h[0] = *(const v8bf*)(pa + kbB);
            a[i].h[1] = *(const v8bf*)(pa + kbB + 8);
        }
#pragma unroll
        for (int j = 0; j < 2; ++j) {
            const __bf16* pb = W + (size_t)(bn0 + wn * 32 + j * 16 + l15) * K + k0 + kbB;
            b[j].h[0] = *(const v8bf*)(pb);
            b[j].h[1] = *(const v8bf*)(pb + 8);
        }
#endif

#pragma unroll
        for (int i = 0; i < 4; ++i)
#pragma unroll
            for (int j = 0; j < 2; ++j)
                acc[i][j] = __builtin_amdgcn_wmma_f32_16x16x32_bf16(
                    false, a[i].v, false, b[j].v, (short)0, acc[i][j], false, false);

#ifdef HAVE_TDM
        __syncthreads();   // all reads of buf 'cur' done before it is refilled
#endif
    }

#pragma unroll
    for (int i = 0; i < 4; ++i)
#pragma unroll
        for (int j = 0; j < 2; ++j)
#pragma unroll
            for (int v = 0; v < 8; ++v) {
                const int m = bm0 + wm * 64 + i * 16 + v + half * 8;
                const int n = bn0 + wn * 32 + j * 16 + l15;
                if (OUT_BF16)
                    ((__bf16*)Cout)[(size_t)m * N + n] = (__bf16)acc[i][j][v];
                else
                    ((float*)Cout)[(size_t)m * N + n] = acc[i][j][v];
            }
    (void)M;
}

// ---------------------------------------------------------------------------
// RoPE + head split + V transpose.
// qkv bf16 [B, L, 3*DM] -> q,k bf16 [B,H,L,HD], vt bf16 [B,H,HD,L]
// ---------------------------------------------------------------------------
__global__ void rope_split(const __bf16* __restrict__ qkv,
                           const float* __restrict__ fcos, const float* __restrict__ fsin,
                           __bf16* __restrict__ q, __bf16* __restrict__ k,
                           __bf16* __restrict__ vt) {
    const int idx = blockIdx.x * 256 + threadIdx.x;      // 2^21 threads
    const int i   = idx & 31;
    const int h   = (idx >> 5) & 15;
    const int pos = (idx >> 9) & 2047;
    const int b   = idx >> 20;

    const __bf16* base = qkv + (size_t)(b * L_SEQ + pos) * (3 * DM);
    const int d0 = h * HD + 2 * i;
    const float qe = (float)base[d0],          qo = (float)base[d0 + 1];
    const float ke = (float)base[DM + d0],     ko = (float)base[DM + d0 + 1];
    const float ve = (float)base[2 * DM + d0], vo = (float)base[2 * DM + d0 + 1];
    const float c  = fcos[pos * 32 + i];
    const float s  = fsin[pos * 32 + i];

    const size_t qk = ((size_t)(b * NH + h) * L_SEQ + pos) * HD + 2 * i;
    q[qk]     = (__bf16)(qe * c - qo * s);
    q[qk + 1] = (__bf16)(qe * s + qo * c);
    k[qk]     = (__bf16)(ke * c - ko * s);
    k[qk + 1] = (__bf16)(ke * s + ko * c);

    const size_t vb = ((size_t)(b * NH + h) * HD + 2 * i) * L_SEQ + pos;
    vt[vb]         = (__bf16)ve;
    vt[vb + L_SEQ] = (__bf16)vo;
}

// ---------------------------------------------------------------------------
// Flash attention: 1 wave = 16 queries x head dim 64, online softmax over
// key chunks of 32. Block = 8 waves = 128 queries; grid = B*H*(L/128) = 512.
// ---------------------------------------------------------------------------
__global__ void __launch_bounds__(256)
flash_attn(const __bf16* __restrict__ Q, const __bf16* __restrict__ K,
           const __bf16* __restrict__ Vt, __bf16* __restrict__ O) {
    __shared__ __align__(32) __bf16 pTile[8][16][32];    // per-wave P staging (8 KB)

    const int lane = threadIdx.x & 31;
    const int wave = threadIdx.x >> 5;
    const int half = lane >> 4;
    const int l15  = lane & 15;
    const int bh   = blockIdx.x >> 4;        // 0..31  (b*NH + h)
    const int qblk = blockIdx.x & 15;
    const int q0   = qblk * 128 + wave * 16;

    const __bf16* Qh = Q  + (size_t)bh * L_SEQ * HD;
    const __bf16* Kh = K  + (size_t)bh * L_SEQ * HD;
    const __bf16* Vh = Vt + (size_t)bh * HD * L_SEQ;
    const int kbA = half * 8;

    AFrag aq[2];
    {
        const __bf16* pq = Qh + (size_t)(q0 + l15) * HD;
#pragma unroll
        for (int c = 0; c < 2; ++c) {
            aq[c].h[0] = *(const v8bf*)(pq + c * 32 + half * 16);
            aq[c].h[1] = *(const v8bf*)(pq + c * 32 + half * 16 + 8);
        }
    }

    v8f acc[4] = {};
    float mrow[8], lrow[8];
#pragma unroll
    for (int v = 0; v < 8; ++v) { mrow[v] = -1e30f; lrow[v] = 0.0f; }

    for (int kt = 0; kt < L_SEQ; kt += 32) {
        __builtin_prefetch(Kh + (size_t)(kt + 32 + l15) * HD, 0, 0);

        // ---- S = scale * Q @ K^T for 2 key sub-tiles of 16 ----
        v8f S[2];
#pragma unroll
        for (int j = 0; j < 2; ++j) {
            AFrag bk0, bk1;
            const __bf16* pk = Kh + (size_t)(kt + j * 16 + l15) * HD + half * 16;
            bk0.h[0] = *(const v8bf*)(pk);
            bk0.h[1] = *(const v8bf*)(pk + 8);
            bk1.h[0] = *(const v8bf*)(pk + 32);
            bk1.h[1] = *(const v8bf*)(pk + 40);
            v8f s = {};
            s = __builtin_amdgcn_wmma_f32_16x16x32_bf16(false, aq[0].v, false, bk0.v, (short)0, s, false, false);
            s = __builtin_amdgcn_wmma_f32_16x16x32_bf16(false, aq[1].v, false, bk1.v, (short)0, s, false, false);
            S[j] = s;
        }

        // ---- online softmax: new row max across the 32 keys ----
        float mnew[8];
#pragma unroll
        for (int v = 0; v < 8; ++v) {
            float s0 = S[0][v] * 0.125f;
            float s1 = S[1][v] * 0.125f;
            S[0][v] = s0; S[1][v] = s1;
            float mx = fmaxf(s0, s1);
            mx = fmaxf(mx, __shfl_xor(mx, 1, 32));
            mx = fmaxf(mx, __shfl_xor(mx, 2, 32));
            mx = fmaxf(mx, __shfl_xor(mx, 4, 32));
            mx = fmaxf(mx, __shfl_xor(mx, 8, 32));
            mnew[v] = fmaxf(mrow[v], mx);
        }
#pragma unroll
        for (int v = 0; v < 8; ++v) {
            const float f = __expf(mrow[v] - mnew[v]);
            lrow[v] *= f;
#pragma unroll
            for (int j = 0; j < 4; ++j) acc[j][v] *= f;
            mrow[v] = mnew[v];
        }

        // ---- P = exp(S - m); row sums; stage P to LDS in [row][key] order ----
#pragma unroll
        for (int j = 0; j < 2; ++j)
#pragma unroll
            for (int v = 0; v < 8; ++v) {
                const float p = __expf(S[j][v] - mrow[v]);
                float ps = p;
                ps += __shfl_xor(ps, 1, 32);
                ps += __shfl_xor(ps, 2, 32);
                ps += __shfl_xor(ps, 4, 32);
                ps += __shfl_xor(ps, 8, 32);
                lrow[v] += ps;
                pTile[wave][v + half * 8][j * 16 + l15] = (__bf16)p;
            }
        __syncthreads();

        // ---- P (A-layout from LDS) @ V (transposed-stored, B contiguous) ----
        AFrag ap;
        {
            const __bf16* pp = &pTile[wave][l15][0];
            ap.h[0] = *(const v8bf*)(pp + half * 16);
            ap.h[1] = *(const v8bf*)(pp + half * 16 + 8);
        }
#pragma unroll
        for (int j = 0; j < 4; ++j) {
            AFrag bv;
            const __bf16* pv = Vh + (size_t)(j * 16 + l15) * L_SEQ + kt + half * 16;
            bv.h[0] = *(const v8bf*)(pv);
            bv.h[1] = *(const v8bf*)(pv + 8);
            acc[j] = __builtin_amdgcn_wmma_f32_16x16x32_bf16(
                false, ap.v, false, bv.v, (short)0, acc[j], false, false);
        }
        __syncthreads();
    }

    // ---- normalize and store to O[B, L, DM] (token-major, bf16) ----
    const int b = bh >> 4;
    const int h = bh & 15;
#pragma unroll
    for (int j = 0; j < 4; ++j)
#pragma unroll
        for (int v = 0; v < 8; ++v) {
            const float o = acc[j][v] / lrow[v];
            const int m = q0 + v + half * 8;
            const size_t off = (size_t)(b * L_SEQ + m) * DM + h * HD + j * 16 + l15;
            O[off] = (__bf16)o;
        }
}

// ---------------------------------------------------------------------------
// launcher
// ---------------------------------------------------------------------------
extern "C" void kernel_launch(void* const* d_in, const int* in_sizes, int n_in,
                              void* d_out, int out_size, void* d_ws, size_t ws_size,
                              hipStream_t stream) {
    (void)in_sizes; (void)n_in; (void)out_size; (void)ws_size;

    const float* hs    = (const float*)d_in[0];   // [2,2048,1024]
    const float* w_qkv = (const float*)d_in[1];   // [3072,1024]
    const float* w_o   = (const float*)d_in[2];   // [1024,1024]
    const float* fcos  = (const float*)d_in[3];   // [2048,32]
    const float* fsin  = (const float*)d_in[4];   // [2048,32]
    float*       out   = (float*)d_out;           // [2,2048,1024]

    char* ws = (char*)d_ws;
    __bf16* hsb  = (__bf16*)(ws + (size_t)( 0 << 20));   //  8 MB [4096,1024]
    __bf16* wqb  = (__bf16*)(ws + (size_t)( 8 << 20));   //  6 MB [3072,1024]
    __bf16* wob  = (__bf16*)(ws + (size_t)(14 << 20));   //  2 MB [1024,1024]
    __bf16* qkvb = (__bf16*)(ws + (size_t)(16 << 20));   // 24 MB [4096,3072]
    __bf16* q    = (__bf16*)(ws + (size_t)(40 << 20));   //  8 MB [B,H,L,HD]
    __bf16* k    = (__bf16*)(ws + (size_t)(48 << 20));   //  8 MB [B,H,L,HD]
    __bf16* vt   = (__bf16*)(ws + (size_t)(56 << 20));   //  8 MB [B,H,HD,L]
    __bf16* o    = (__bf16*)(ws + (size_t)(64 << 20));   //  8 MB [4096,1024]

    const int n1 = B_SZ * L_SEQ * DM;   // 4194304
    const int n2 = 3 * DM * DM;         // 3145728
    const int n3 = DM * DM;             // 1048576
    f32_to_bf16<<<(n1 + 255) / 256, 256, 0, stream>>>(hs,    hsb, n1);
    f32_to_bf16<<<(n2 + 255) / 256, 256, 0, stream>>>(w_qkv, wqb, n2);
    f32_to_bf16<<<(n3 + 255) / 256, 256, 0, stream>>>(w_o,   wob, n3);

    // QKV projection: [4096,1024] x [3072,1024]^T -> bf16 [4096,3072]
    dim3 g1(3 * DM / 128, B_SZ * L_SEQ / 128);   // (24, 32)
    wmma_gemm<true><<<g1, 256, 0, stream>>>(hsb, wqb, (void*)qkvb,
                                            B_SZ * L_SEQ, 3 * DM, DM);

    // RoPE + split + V transpose
    const int n4 = B_SZ * L_SEQ * NH * (HD / 2);  // 2097152
    rope_split<<<n4 / 256, 256, 0, stream>>>(qkvb, fcos, fsin, q, k, vt);

    // attention: 512 blocks x 256 threads
    flash_attn<<<B_SZ * NH * (L_SEQ / 128), 256, 0, stream>>>(q, k, vt, o);

    // output projection: [4096,1024] x [1024,1024]^T -> f32 out
    dim3 g2(DM / 128, B_SZ * L_SEQ / 128);       // (8, 32)
    wmma_gemm<false><<<g2, 256, 0, stream>>>(o, wob, (void*)out,
                                             B_SZ * L_SEQ, DM, DM);
}
